// GAT_80814104642288
// MI455X (gfx1250) — matrix operand order repaired and checked by reading, not compiled, not run on hardware
//
#include <hip/hip_runtime.h>
#include <hip/hip_bf16.h>

// ---------------------------------------------------------------- types
typedef __attribute__((ext_vector_type(16))) _Float16 v16h;
typedef __attribute__((ext_vector_type(8)))  _Float16 v8h;
typedef __attribute__((ext_vector_type(8)))  float    v8f;

#define TPB 256

// ---------------------------------------------------------------- helpers
__device__ __forceinline__ unsigned f2o(float f) {
  unsigned u = __float_as_uint(f);
  return (u & 0x80000000u) ? ~u : (u | 0x80000000u);   // order-preserving map
}
__device__ __forceinline__ float o2f(unsigned o) {
  unsigned u = (o & 0x80000000u) ? (o & 0x7FFFFFFFu) : ~o;
  return __uint_as_float(u);
}
__device__ __forceinline__ void edge_sd(const int* __restrict__ eg, int E, int idx,
                                        int& s, int& d) {
  if (idx < E) { s = eg[idx]; d = eg[E + idx]; }
  else         { s = d = idx - E; }                    // self loop
}

// ---------------------------------------------------------------- zero fill
__global__ void gat_zero_u32(unsigned* __restrict__ p, int n) {
  int i = blockIdx.x * TPB + threadIdx.x;
  if (i < n) p[i] = 0u;
}

// ---------------------------------------------------------------- f32 -> f16
__global__ void gat_cvt_half(const float* __restrict__ in, _Float16* __restrict__ out, int n) {
  int i = blockIdx.x * TPB + threadIdx.x;
  if (i < n) out[i] = (_Float16)in[i];
}

// ---------------------------------------------------------------- layer-1 GEMM via WMMA
// C[n, j] = sum_k A[n,k] * W1[j,k]   (A: [N,128] f16, B=W1: [128,128] f16, C: [N,128] f32)
// One wave computes one 16x16 tile of C; K loop = 4 x wmma_f32_16x16x32_f16.
__global__ void __launch_bounds__(TPB)
gat_gemm1_wmma(const _Float16* __restrict__ A, const _Float16* __restrict__ B,
               float* __restrict__ C, int nrows) {
  const int lane   = threadIdx.x & 31;
  const int wave   = blockIdx.x * (TPB / 32) + (threadIdx.x >> 5);
  const int tileM  = wave >> 3;       // 8 N-tiles (128/16) per M row
  const int tileN  = wave & 7;
  if (tileM * 16 >= nrows) return;    // uniform per wave

  const int l15  = lane & 15;
  const int hiA  = (lane & 16) ? 16 : 0;   // contiguous K order (matches B)
  const int hiB  = (lane & 16) ? 16 : 0;   // contiguous K order (matches A)
  const _Float16* arow = A + (size_t)(tileM * 16 + l15) * 128;
  const _Float16* brow = B + (size_t)(tileN * 16 + l15) * 128;

  v8f acc = {};
#pragma unroll
  for (int kb = 0; kb < 4; ++kb) {
    const int k0 = kb * 32;
    union { v16h v; v8h h[2]; } ua;
    ua.h[0] = *(const v8h*)(arow + k0 + hiA);        // K = k0+hiA   .. +7
    ua.h[1] = *(const v8h*)(arow + k0 + 8 + hiA);    // K = k0+8+hiA .. +7
    v16h bf = *(const v16h*)(brow + k0 + hiB);       // K = k0+hiB   .. +15
    acc = __builtin_amdgcn_wmma_f32_16x16x32_f16(
        /*neg_a=*/false, ua.v, /*neg_b=*/false, bf,
        /*c_mod=*/(short)0, acc, /*reuse_a=*/false, /*reuse_b=*/false);
  }

  const int mAdd = (lane & 16) ? 8 : 0;              // C/D layout: VGPR r -> M=r(+8)
  float* crow = C + (size_t)(tileM * 16) * 128 + tileN * 16 + l15;
#pragma unroll
  for (int r = 0; r < 8; ++r)
    crow[(size_t)(r + mAdd) * 128] = acc[r];
}

// ---------------------------------------------------------------- attention coefficients L1
// alpha_s[n,h] = sum_c xp[n,h,c]*a_src[h,c] ; same for dst. H=2, C=64.
__global__ void gat_attn1(const float* __restrict__ xp,
                          const float* __restrict__ a_src, const float* __restrict__ a_dst,
                          float* __restrict__ as, float* __restrict__ ad, int n2) {
  int i = blockIdx.x * TPB + threadIdx.x;
  if (i >= n2) return;
  int node = i >> 1, h = i & 1;
  const float* v = xp + (size_t)node * 128 + h * 64;
  const float* s = a_src + h * 64;
  const float* d = a_dst + h * 64;
  float accs = 0.f, accd = 0.f;
#pragma unroll 8
  for (int c = 0; c < 64; ++c) { float x = v[c]; accs += x * s[c]; accd += x * d[c]; }
  as[i] = accs; ad[i] = accd;
}

// ---------------------------------------------------------------- L1 edge logits + segment max
__global__ void gat_logits1(const int* __restrict__ eg, int E, int Et2,
                            const float* __restrict__ as, const float* __restrict__ ad,
                            float* __restrict__ e1, unsigned* __restrict__ m1) {
  int i = blockIdx.x * TPB + threadIdx.x;
  if (i >= Et2) return;
  int e = i >> 1, h = i & 1, s, d;
  edge_sd(eg, E, e, s, d);
  float v = as[s * 2 + h] + ad[d * 2 + h];
  v = v >= 0.f ? v : 0.2f * v;                      // leaky relu
  e1[i] = v;
  atomicMax(&m1[d * 2 + h], f2o(v));
}

// ---------------------------------------------------------------- L1 exp + segment sum
__global__ void gat_expsum1(const int* __restrict__ eg, int E, int Et2,
                            float* __restrict__ e1, const unsigned* __restrict__ m1,
                            float* __restrict__ s1) {
  int i = blockIdx.x * TPB + threadIdx.x;
  if (i >= Et2) return;
  int e = i >> 1, h = i & 1, s, d;
  edge_sd(eg, E, e, s, d);
  float ex = __expf(e1[i] - o2f(m1[d * 2 + h]));
  e1[i] = ex;
  atomicAdd(&s1[d * 2 + h], ex);
}

// ---------------------------------------------------------------- L1 normalize
__global__ void gat_norm1(const int* __restrict__ eg, int E, int Et2,
                          float* __restrict__ e1, const float* __restrict__ s1) {
  int i = blockIdx.x * TPB + threadIdx.x;
  if (i >= Et2) return;
  int e = i >> 1, h = i & 1, s, d;
  edge_sd(eg, E, e, s, d);
  e1[i] = e1[i] / s1[d * 2 + h];
}

// ---------------------------------------------------------------- L1 scatter aggregation (hot)
__global__ void gat_agg1(const int* __restrict__ eg, int E, int total,
                         const float* __restrict__ alpha, const float* __restrict__ xp,
                         float* __restrict__ agg) {
  int i = blockIdx.x * TPB + threadIdx.x;
  if (i >= total) return;
  int e = i >> 7, c = i & 127, h = c >> 6, s, d;
  edge_sd(eg, E, e, s, d);
  atomicAdd(&agg[(size_t)d * 128 + c], alpha[e * 2 + h] * xp[(size_t)s * 128 + c]);
}

// ---------------------------------------------------------------- L1 bias + relu -> h
__global__ void gat_final1(const float* __restrict__ agg, const float* __restrict__ b1,
                           float* __restrict__ h, int total) {
  int i = blockIdx.x * TPB + threadIdx.x;
  if (i >= total) return;
  float v = agg[i] + b1[i & 127];
  h[i] = v > 0.f ? v : 0.f;
}

// ---------------------------------------------------------------- fused mu/std projection (Z=2)
__global__ void gat_proj23(const float* __restrict__ h,
                           const float* __restrict__ Wmu, const float* __restrict__ Wsd,
                           const float* __restrict__ asmu_v, const float* __restrict__ admu_v,
                           const float* __restrict__ assd_v, const float* __restrict__ adsd_v,
                           float* __restrict__ xpmu, float* __restrict__ xpsd,
                           float* __restrict__ asmu, float* __restrict__ admu,
                           float* __restrict__ assd, float* __restrict__ adsd, int n) {
  int i = blockIdx.x * TPB + threadIdx.x;
  if (i >= n) return;
  const float* hv = h + (size_t)i * 128;
  float m0 = 0.f, m1 = 0.f, s0 = 0.f, s1 = 0.f;
#pragma unroll 4
  for (int j = 0; j < 128; ++j) {
    float x = hv[j];
    m0 += x * Wmu[j];        m1 += x * Wmu[128 + j];
    s0 += x * Wsd[j];        s1 += x * Wsd[128 + j];
  }
  xpmu[i * 2 + 0] = m0;  xpmu[i * 2 + 1] = m1;
  xpsd[i * 2 + 0] = s0;  xpsd[i * 2 + 1] = s1;
  asmu[i] = m0 * asmu_v[0] + m1 * asmu_v[1];
  admu[i] = m0 * admu_v[0] + m1 * admu_v[1];
  assd[i] = s0 * assd_v[0] + s1 * assd_v[1];
  adsd[i] = s0 * adsd_v[0] + s1 * adsd_v[1];
}

// ---------------------------------------------------------------- L2/L3 edge pipeline (H=1, fused)
__global__ void gat_logits23(const int* __restrict__ eg, int E, int Et,
                             const float* __restrict__ asmu, const float* __restrict__ admu,
                             const float* __restrict__ assd, const float* __restrict__ adsd,
                             float* __restrict__ emu, float* __restrict__ esd,
                             unsigned* __restrict__ mmu, unsigned* __restrict__ msd) {
  int i = blockIdx.x * TPB + threadIdx.x;
  if (i >= Et) return;
  int s, d;
  edge_sd(eg, E, i, s, d);
  float vm = asmu[s] + admu[d]; vm = vm >= 0.f ? vm : 0.2f * vm;
  float vs = assd[s] + adsd[d]; vs = vs >= 0.f ? vs : 0.2f * vs;
  emu[i] = vm; esd[i] = vs;
  atomicMax(&mmu[d], f2o(vm));
  atomicMax(&msd[d], f2o(vs));
}

__global__ void gat_expsum23(const int* __restrict__ eg, int E, int Et,
                             float* __restrict__ emu, float* __restrict__ esd,
                             const unsigned* __restrict__ mmu, const unsigned* __restrict__ msd,
                             float* __restrict__ smu, float* __restrict__ ssd) {
  int i = blockIdx.x * TPB + threadIdx.x;
  if (i >= Et) return;
  int s, d;
  edge_sd(eg, E, i, s, d);
  float xm = __expf(emu[i] - o2f(mmu[d]));
  float xs = __expf(esd[i] - o2f(msd[d]));
  emu[i] = xm; esd[i] = xs;
  atomicAdd(&smu[d], xm);
  atomicAdd(&ssd[d], xs);
}

__global__ void gat_agg23(const int* __restrict__ eg, int E, int Et,
                          const float* __restrict__ emu, const float* __restrict__ esd,
                          const float* __restrict__ smu, const float* __restrict__ ssd,
                          const float* __restrict__ xpmu, const float* __restrict__ xpsd,
                          float* __restrict__ aggmu, float* __restrict__ aggsd) {
  int i = blockIdx.x * TPB + threadIdx.x;
  if (i >= Et) return;
  int s, d;
  edge_sd(eg, E, i, s, d);
  float am = emu[i] / smu[d];
  float as = esd[i] / ssd[d];
  atomicAdd(&aggmu[d * 2 + 0], am * xpmu[s * 2 + 0]);
  atomicAdd(&aggmu[d * 2 + 1], am * xpmu[s * 2 + 1]);
  atomicAdd(&aggsd[d * 2 + 0], as * xpsd[s * 2 + 0]);
  atomicAdd(&aggsd[d * 2 + 1], as * xpsd[s * 2 + 1]);
}

__global__ void gat_final23(const float* __restrict__ aggmu, const float* __restrict__ aggsd,
                            const float* __restrict__ bmu, const float* __restrict__ bsd,
                            float* __restrict__ out, int n2) {
  int i = blockIdx.x * TPB + threadIdx.x;
  if (i >= n2) return;
  out[i]      = aggmu[i] + bmu[i & 1];
  out[n2 + i] = aggsd[i] + bsd[i & 1];
}

// ================================================================ host
static inline int blks(long long n) { return (int)((n + TPB - 1) / TPB); }

extern "C" void kernel_launch(void* const* d_in, const int* in_sizes, int n_in,
                              void* d_out, int out_size, void* d_ws, size_t ws_size,
                              hipStream_t stream) {
  const float* features  = (const float*)d_in[0];
  const int*   edges     = (const int*)  d_in[1];
  const float* W1        = (const float*)d_in[2];
  const float* a_src1    = (const float*)d_in[3];
  const float* a_dst1    = (const float*)d_in[4];
  const float* b1        = (const float*)d_in[5];
  const float* W_mu      = (const float*)d_in[6];
  const float* a_src_mu  = (const float*)d_in[7];
  const float* a_dst_mu  = (const float*)d_in[8];
  const float* b_mu      = (const float*)d_in[9];
  const float* W_std     = (const float*)d_in[10];
  const float* a_src_std = (const float*)d_in[11];
  const float* a_dst_std = (const float*)d_in[12];
  const float* b_std     = (const float*)d_in[13];
  float* out = (float*)d_out;

  const int FN = 128, HC = 128;
  const int N  = in_sizes[0] / FN;    // 50000
  const int E  = in_sizes[1] / 2;     // 800000
  const int Et = E + N;               // with self loops

  // -------- workspace carve-out (all offsets 256B aligned)
  char*  base = (char*)d_ws;
  size_t off  = 0;
  auto carve = [&](size_t bytes) -> char* {
    char* p = base + off;
    off += (bytes + 255) & ~(size_t)255;
    return p;
  };
  // zeroed region (contiguous, cleared every call)
  char*     zbase  = base + off;
  float*    agg1   = (float*)   carve((size_t)N * HC * 4);
  unsigned* m1     = (unsigned*)carve((size_t)N * 2 * 4);
  float*    s1     = (float*)   carve((size_t)N * 2 * 4);
  unsigned* mmu    = (unsigned*)carve((size_t)N * 4);
  float*    smu    = (float*)   carve((size_t)N * 4);
  unsigned* msd    = (unsigned*)carve((size_t)N * 4);
  float*    ssd    = (float*)   carve((size_t)N * 4);
  float*    aggmu  = (float*)   carve((size_t)N * 2 * 4);
  float*    aggsd  = (float*)   carve((size_t)N * 2 * 4);
  size_t    zbytes = (size_t)((base + off) - zbase);
  // live scratch
  _Float16* feat16 = (_Float16*)carve((size_t)N * FN * 2);
  _Float16* w16    = (_Float16*)carve((size_t)HC * FN * 2);
  float*    xp1    = (float*)   carve((size_t)N * HC * 4);   // reused as h after agg
  float*    as1    = (float*)   carve((size_t)N * 2 * 4);
  float*    ad1    = (float*)   carve((size_t)N * 2 * 4);
  float*    e1     = (float*)   carve((size_t)Et * 2 * 4);
  float*    xpmu   = (float*)   carve((size_t)N * 2 * 4);
  float*    xpsd   = (float*)   carve((size_t)N * 2 * 4);
  float*    asmu   = (float*)   carve((size_t)N * 4);
  float*    admu   = (float*)   carve((size_t)N * 4);
  float*    assd   = (float*)   carve((size_t)N * 4);
  float*    adsd   = (float*)   carve((size_t)N * 4);
  float*    emu    = (float*)   carve((size_t)Et * 4);
  float*    esd    = (float*)   carve((size_t)Et * 4);
  (void)ws_size;

  // -------- 0) clear accumulators / segment-max (encoded 0 == -inf)
  gat_zero_u32<<<blks(zbytes / 4), TPB, 0, stream>>>((unsigned*)zbase, (int)(zbytes / 4));

  // -------- 1) convert inputs to f16 for WMMA
  gat_cvt_half<<<blks((long long)N * FN), TPB, 0, stream>>>(features, feat16, N * FN);
  gat_cvt_half<<<blks(HC * FN), TPB, 0, stream>>>(W1, w16, HC * FN);

  // -------- 2) layer-1 dense projection: xp1 = x @ W1^T   (WMMA)
  {
    int tilesM = (N + 15) / 16;                 // 3125 (exact)
    int waves  = tilesM * (HC / 16);            // 25000
    gat_gemm1_wmma<<<(waves + 7) / 8, TPB, 0, stream>>>(feat16, w16, xp1, N);
  }

  // -------- 3) attention coefficients
  gat_attn1<<<blks((long long)N * 2), TPB, 0, stream>>>(xp1, a_src1, a_dst1, as1, ad1, N * 2);

  // -------- 4) layer-1 edge softmax + aggregation
  gat_logits1<<<blks((long long)Et * 2), TPB, 0, stream>>>(edges, E, Et * 2, as1, ad1, e1, m1);
  gat_expsum1<<<blks((long long)Et * 2), TPB, 0, stream>>>(edges, E, Et * 2, e1, m1, s1);
  gat_norm1 <<<blks((long long)Et * 2), TPB, 0, stream>>>(edges, E, Et * 2, e1, s1);
  gat_agg1  <<<blks((long long)Et * 128), TPB, 0, stream>>>(edges, E, Et * 128, e1, xp1, agg1);
  gat_final1<<<blks((long long)N * HC), TPB, 0, stream>>>(agg1, b1, xp1 /* h in-place */, N * HC);

  // -------- 5) fused mu/std projection + attention coefficients (Z=2)
  gat_proj23<<<blks(N), TPB, 0, stream>>>(xp1, W_mu, W_std,
                                          a_src_mu, a_dst_mu, a_src_std, a_dst_std,
                                          xpmu, xpsd, asmu, admu, assd, adsd, N);

  // -------- 6) layer-2/3 edge softmax + aggregation (fused)
  gat_logits23<<<blks(Et), TPB, 0, stream>>>(edges, E, Et, asmu, admu, assd, adsd,
                                             emu, esd, mmu, msd);
  gat_expsum23<<<blks(Et), TPB, 0, stream>>>(edges, E, Et, emu, esd, mmu, msd, smu, ssd);
  gat_agg23   <<<blks(Et), TPB, 0, stream>>>(edges, E, Et, emu, esd, smu, ssd,
                                             xpmu, xpsd, aggmu, aggsd);

  // -------- 7) bias add -> d_out  (mu flat [N,2], then std flat [N,2])
  gat_final23<<<blks((long long)N * 2), TPB, 0, stream>>>(aggmu, aggsd, b_mu, b_std, out, N * 2);
}
